// ManyToMany_8220567404556
// MI455X (gfx1250) — matrix operand order repaired and checked by reading, not compiled, not run on hardware
//
#include <hip/hip_runtime.h>
#include <stdint.h>

// ---------------------------------------------------------------------------
// Seq2seq LSTM + attention for MI455X (gfx1250, wave32, WMMA bf16).
// All GEMMs run through v_wmma_f32_16x16x32_bf16 with f32 accumulation.
// The dominant output-projection GEMM stages its A panel in LDS via the
// Tensor Data Mover (TENSOR_LOAD_TO_LDS + s_wait_tensorcnt) and computes two
// N tiles per wave to raise arithmetic intensity.
// ---------------------------------------------------------------------------

#define DI __device__ __forceinline__

typedef __attribute__((ext_vector_type(16))) __bf16 v16bf;
typedef __attribute__((ext_vector_type(8)))  float  v8f;
typedef __attribute__((ext_vector_type(4)))  unsigned int u32x4;
typedef __attribute__((ext_vector_type(8)))  int i32x8;
typedef __attribute__((ext_vector_type(4)))  int i32x4;

// Problem constants
#define BB   32          // batch
#define SS   128         // encoder seq len
#define TT   32          // decoder steps (T-1)
#define EE   512         // embed dim
#define HH   512         // hidden
#define VD_  32000       // vocab

// ---------------- WMMA fragment helpers (ISA 7.12.2 layouts) ----------------

// A 16x32 bf16 tile, row-major source with leading dim lda.
// lanes 0-15: M=lane, K = {0..7, 16..23}; lanes 16-31: M=lane-16, K={8..15,24..31}
DI v16bf load_a(const __bf16* A, long lda, int k0, int lane) {
  int m  = lane & 15;
  int kb = (lane >> 4) * 8;
  const __bf16* p = A + (long)m * lda + k0 + kb;
  v16bf a;
#pragma unroll
  for (int j = 0; j < 8; ++j) a[j] = p[j];
#pragma unroll
  for (int j = 0; j < 8; ++j) a[j + 8] = p[16 + j];
  return a;
}

// B 32x16 bf16 tile: N = lane%16; lanes 0-15 K = {0..7, 16..23}; lanes 16-31 K = {8..15, 24..31}.
DI v16bf load_b(const __bf16* B, long sK, long sN, int k0, int lane) {
  const __bf16* p = B + (long)(k0 + (lane >> 4) * 8) * sK + (long)(lane & 15) * sN;
  v16bf b;
#pragma unroll
  for (int j = 0; j < 8; ++j) b[j] = p[(long)j * sK];
#pragma unroll
  for (int j = 0; j < 8; ++j) b[j + 8] = p[(long)(16 + j) * sK];
  return b;
}

DI v8f wmma_bf16(v8f acc, v16bf a, v16bf b) {
  return __builtin_amdgcn_wmma_f32_16x16x32_bf16(false, a, false, b,
                                                 (short)0, acc, false, false);
}

DI v8f gemm16(v8f acc, const __bf16* A, long lda,
              const __bf16* B, long sK, long sN, int K, int lane) {
  for (int k = 0; k < K; k += 32)
    acc = wmma_bf16(acc, load_a(A, lda, k, lane), load_b(B, sK, sN, k, lane));
  return acc;
}

DI float sigmoidf_(float x) { return 1.0f / (1.0f + __expf(-x)); }

// ---------------- weight convert + transpose (f32 [R,C] -> bf16 [C,R]) -----

__global__ void k_conv_transpose(const float* __restrict__ src,
                                 __bf16* __restrict__ dst, int R, int C) {
  __shared__ float tile[32][33];
  int c0 = blockIdx.x * 32, r0 = blockIdx.y * 32;
  for (int i = threadIdx.y; i < 32; i += 8) {
    int r = r0 + i, c = c0 + threadIdx.x;
    tile[i][threadIdx.x] = (r < R && c < C) ? src[(long)r * C + c] : 0.0f;
  }
  __syncthreads();
  for (int i = threadIdx.y; i < 32; i += 8) {
    int c = c0 + i, r = r0 + threadIdx.x;   // dst[c][r]
    if (c < C && r < R) dst[(long)c * R + r] = (__bf16)tile[threadIdx.x][i];
  }
}

// ---------------- embedding gathers -----------------------------------------

__global__ void k_gather_enc(const int* __restrict__ x,
                             const float* __restrict__ emb,
                             __bf16* __restrict__ ex) {        // ex[s][b][e]
  int sb = blockIdx.x;            // s*32 + b
  int s = sb >> 5, b = sb & 31;
  int tok = x[b * SS + s];
  const float* src = emb + (long)tok * EE;
  __bf16* dst = ex + (long)sb * EE;
  for (int e = threadIdx.x; e < EE; e += 256) dst[e] = (__bf16)src[e];
}

__global__ void k_gather_dec(const int* __restrict__ x, const int* __restrict__ y,
                             const float* __restrict__ emb,
                             __bf16* __restrict__ Adec) {      // left half of [1024][1024]
  int r = blockIdx.x;             // b*32 + t
  int b = r >> 5, t = r & 31;
  int tok = (t == 0) ? x[b * SS] : y[b * 33 + t];
  const float* src = emb + (long)tok * EE;
  __bf16* dst = Adec + (long)r * 1024;
  for (int e = threadIdx.x; e < EE; e += 256) dst[e] = (__bf16)src[e];
}

__global__ void k_fill_dec_h(const __bf16* __restrict__ hfin,
                             __bf16* __restrict__ Adec) {      // right half
  int r = blockIdx.x; int b = r >> 5;
  const __bf16* src = hfin + (long)b * HH;
  __bf16* dst = Adec + (long)r * 1024 + EE;
  for (int e = threadIdx.x; e < HH; e += 256) dst[e] = src[e];
}

__global__ void k_init_state(__bf16* h0, float* c0) {
  int i = blockIdx.x * 256 + threadIdx.x;
  if (i < BB * HH) { h0[i] = (__bf16)0.0f; c0[i] = 0.0f; }
}

// ---------------- encoder: one recurrent step --------------------------------
// grid (32 n-tiles, 2 m-tiles), block 128 = 4 waves (f, i, o, cand)

__global__ void k_enc_step(const __bf16* __restrict__ ex_t,   // [32][512]
                           const __bf16* __restrict__ h_in,   // [32][512] bf16
                           const float*  __restrict__ c_in,   // [32][512] f32
                           const __bf16* WfT, const __bf16* WiT,
                           const __bf16* WoT, const __bf16* WcT, // [1024][512]
                           const float* bf_, const float* bi_,
                           const float* bo_, const float* bc_,
                           __bf16* __restrict__ h_out, float* __restrict__ c_out,
                           __bf16* __restrict__ hs_t) {
  __shared__ float gsh[4][16][16];
  int lane = threadIdx.x & 31;
  int wave = threadIdx.x >> 5;                 // gate id: f,i,o,c
  int n0 = blockIdx.x * 16;
  int m0 = blockIdx.y * 16;
  const __bf16* WT  = (wave == 0) ? WfT : (wave == 1) ? WiT : (wave == 2) ? WoT : WcT;
  const float* bias = (wave == 0) ? bf_ : (wave == 1) ? bi_ : (wave == 2) ? bo_ : bc_;

  v8f acc = {0.f,0.f,0.f,0.f,0.f,0.f,0.f,0.f};
  // K = [emb 0..511 | h 512..1023]
  acc = gemm16(acc, ex_t + (long)m0 * EE, EE, WT + n0,                 HH, 1, EE, lane);
  acc = gemm16(acc, h_in + (long)m0 * HH, HH, WT + (long)EE * HH + n0, HH, 1, HH, lane);

  int n  = lane & 15;
  int mb = (lane >> 4) * 8;
  float bn = bias[n0 + n];
#pragma unroll
  for (int r = 0; r < 8; ++r) gsh[wave][mb + r][n] = acc[r] + bn;
  __syncthreads();

  for (int e = threadIdx.x; e < 256; e += 128) {
    int m = e >> 4, nn = e & 15;
    float f    = sigmoidf_(gsh[0][m][nn]);
    float i    = sigmoidf_(gsh[1][m][nn]);
    float o    = sigmoidf_(gsh[2][m][nn]);
    float cand = tanhf(gsh[3][m][nn]);
    long idx = (long)(m0 + m) * HH + n0 + nn;
    float c = f * c_in[idx] + i * cand;
    float h = o * tanhf(c);
    c_out[idx] = c;
    __bf16 hb = (__bf16)h;
    h_out[idx] = hb;
    hs_t[idx]  = hb;
  }
}

// ---------------- decoder: all 32 steps in parallel --------------------------
// grid (32 n-tiles, 64 m-tiles), block 128 = 4 waves

__global__ void k_dec_step(const __bf16* __restrict__ Adec,   // [1024][1024]
                           const float*  __restrict__ c_enc,  // [32][512]
                           const __bf16* WfT, const __bf16* WiT,
                           const __bf16* WoT, const __bf16* WcT, // [1024][512]
                           const float* bf_, const float* bi_,
                           const float* bo_, const float* bc_,
                           __bf16* __restrict__ dec_h) {      // [1024][512]
  __shared__ float gsh[4][16][16];
  int lane = threadIdx.x & 31;
  int wave = threadIdx.x >> 5;
  int n0 = blockIdx.x * 16;
  int m0 = blockIdx.y * 16;
  const __bf16* WT  = (wave == 0) ? WfT : (wave == 1) ? WiT : (wave == 2) ? WoT : WcT;
  const float* bias = (wave == 0) ? bf_ : (wave == 1) ? bi_ : (wave == 2) ? bo_ : bc_;

  v8f acc = {0.f,0.f,0.f,0.f,0.f,0.f,0.f,0.f};
  acc = gemm16(acc, Adec + (long)m0 * 1024, 1024, WT + n0, HH, 1, 1024, lane);

  int n  = lane & 15;
  int mb = (lane >> 4) * 8;
  float bn = bias[n0 + n];
#pragma unroll
  for (int r = 0; r < 8; ++r) gsh[wave][mb + r][n] = acc[r] + bn;
  __syncthreads();

  for (int e = threadIdx.x; e < 256; e += 128) {
    int m = e >> 4, nn = e & 15;
    float f    = sigmoidf_(gsh[0][m][nn]);
    float i    = sigmoidf_(gsh[1][m][nn]);
    float o    = sigmoidf_(gsh[2][m][nn]);
    float cand = tanhf(gsh[3][m][nn]);
    int row = m0 + m;                          // b*32 + t
    float cprev = c_enc[(long)(row >> 5) * HH + n0 + nn];
    float c = f * cprev + i * cand;
    dec_h[(long)row * HH + n0 + nn] = (__bf16)(o * tanhf(c));
  }
}

// ---------------- attention projection: xl = dec_h @ attW^T + b -------------
// grid (8, 64), block 128.  Writes bf16 into comb[:, 0:512].

__global__ void k_att_proj(const __bf16* __restrict__ dec_h,
                           const __bf16* __restrict__ attWT,  // [512][512]
                           const float* __restrict__ att_b,
                           __bf16* __restrict__ comb) {       // [1024][1024]
  int lane = threadIdx.x & 31;
  int wave = threadIdx.x >> 5;
  int n0 = (blockIdx.x * 4 + wave) * 16;
  int m0 = blockIdx.y * 16;
  v8f acc = {0.f,0.f,0.f,0.f,0.f,0.f,0.f,0.f};
  acc = gemm16(acc, dec_h + (long)m0 * HH, HH, attWT + n0, HH, 1, HH, lane);
  int n = n0 + (lane & 15);
  int mrow = m0 + (lane >> 4) * 8;
  float bn = att_b[n];
#pragma unroll
  for (int r = 0; r < 8; ++r)
    comb[(long)(mrow + r) * 1024 + n] = (__bf16)(acc[r] + bn);
}

// ---------------- attention scores: s[b][t][s'] = xl . enc_hs ---------------
// grid (2, 2, 32 batches), block 128

__global__ void k_scores(const __bf16* __restrict__ comb,     // xl in cols 0..511
                         const __bf16* __restrict__ enc_hs,   // [S][B][H]
                         float* __restrict__ scores) {        // [B][T][S]
  int lane = threadIdx.x & 31;
  int wave = threadIdx.x >> 5;
  int b  = blockIdx.z;
  int n0 = (blockIdx.x * 4 + wave) * 16;       // over S = 128
  int m0 = blockIdx.y * 16;                    // over T = 32
  v8f acc = {0.f,0.f,0.f,0.f,0.f,0.f,0.f,0.f};
  // B[k][n] = enc_hs[n][b][k] : strideK=1, strideN=B*H
  acc = gemm16(acc, comb + (long)(b * TT + m0) * 1024, 1024,
               enc_hs + (long)b * HH + (long)n0 * (BB * HH), 1, BB * HH, HH, lane);
  int n = n0 + (lane & 15);
  int mrow = m0 + (lane >> 4) * 8;
#pragma unroll
  for (int r = 0; r < 8; ++r)
    scores[((long)b * TT + mrow + r) * SS + n] = acc[r];
}

// ---------------- softmax over S (=128) -------------------------------------

__global__ void k_softmax(const float* __restrict__ scores,
                          __bf16* __restrict__ w) {           // [B][T][S] bf16
  int lane = threadIdx.x & 31;
  int wave = threadIdx.x >> 5;
  int row = blockIdx.x * 4 + wave;             // b*32+t
  const float* src = scores + (long)row * SS;
  float v[4];
  float mx = -1e30f;
#pragma unroll
  for (int j = 0; j < 4; ++j) { v[j] = src[lane + 32 * j]; mx = fmaxf(mx, v[j]); }
  for (int off = 16; off; off >>= 1) mx = fmaxf(mx, __shfl_xor(mx, off, 32));
  float sum = 0.f;
#pragma unroll
  for (int j = 0; j < 4; ++j) { v[j] = __expf(v[j] - mx); sum += v[j]; }
  for (int off = 16; off; off >>= 1) sum += __shfl_xor(sum, off, 32);
  float inv = 1.0f / sum;
  __bf16* dst = w + (long)row * SS;
#pragma unroll
  for (int j = 0; j < 4; ++j) dst[lane + 32 * j] = (__bf16)(v[j] * inv);
}

// ---------------- context: ctx = w @ enc_hs ----------------------------------
// grid (8, 2, 32), block 128.  Writes bf16 into comb[:, 512:1024].

__global__ void k_ctx(const __bf16* __restrict__ w,           // [B][T][S]
                      const __bf16* __restrict__ enc_hs,      // [S][B][H]
                      __bf16* __restrict__ comb) {
  int lane = threadIdx.x & 31;
  int wave = threadIdx.x >> 5;
  int b  = blockIdx.z;
  int n0 = (blockIdx.x * 4 + wave) * 16;       // over H
  int m0 = blockIdx.y * 16;                    // over T
  v8f acc = {0.f,0.f,0.f,0.f,0.f,0.f,0.f,0.f};
  // B[k][n] = enc_hs[k][b][n] : strideK=B*H, strideN=1
  acc = gemm16(acc, w + (long)(b * TT + m0) * SS, SS,
               enc_hs + (long)b * HH + n0, BB * HH, 1, SS, lane);
  int n = n0 + (lane & 15);
  int mrow = m0 + (lane >> 4) * 8;
#pragma unroll
  for (int r = 0; r < 8; ++r)
    comb[(long)(b * TT + mrow + r) * 1024 + EE + n] = (__bf16)acc[r];
}

// ---------------- output projection: tanh(comb @ outW^T + b) ----------------
// grid (125, 64), block 256 = 8 waves.  Dominant GEMM: K=1024, N=32000.
// A panel (16x128 bf16) staged in LDS via the Tensor Data Mover; each wave
// computes TWO adjacent N tiles reusing one A fragment.  Stores [T][B][V].

__global__ void k_logits(const __bf16* __restrict__ comb,     // [1024][1024]
                         const __bf16* __restrict__ outWT,    // [1024][32000]
                         const float* __restrict__ out_b,
                         float* __restrict__ out) {           // [T][B][V]
  __shared__ __align__(16) __bf16 shA[16][128];               // 4 KB A panel
  int lane = threadIdx.x & 31;
  int wave = threadIdx.x >> 5;
  int m0 = blockIdx.y * 16;                                   // rows (b*32+t)
  int nbase = (blockIdx.x * 16 + wave * 2) * 16;              // 2 N tiles/wave
  const __bf16* B0 = outWT + nbase;

  v8f acc0 = {0.f,0.f,0.f,0.f,0.f,0.f,0.f,0.f};
  v8f acc1 = {0.f,0.f,0.f,0.f,0.f,0.f,0.f,0.f};

  for (int kc = 0; kc < 1024; kc += 128) {
    __syncthreads();                         // previous panel fully consumed
#if __has_builtin(__builtin_amdgcn_tensor_load_to_lds)
    if (wave == 0) {
      // D# for a 2D 16x128 bf16 tile of comb (row stride 1024 elements)
      uint64_t ga = (uint64_t)(uintptr_t)(comb + (size_t)m0 * 1024 + kc);
      unsigned la = (unsigned)(uintptr_t)(&shA[0][0]);        // LDS offset = addr[31:0]
      u32x4 g0;
      g0[0] = 1u;                                             // count=1, user mode
      g0[1] = la;                                             // lds_addr
      g0[2] = (unsigned)(ga & 0xffffffffu);                   // global_addr[31:0]
      g0[3] = (unsigned)((ga >> 32) & 0x01ffffffu) | 0x80000000u; // addr[56:32]|type=2
      i32x8 g1;
      g1[0] = 0x00010000;        // wg_mask=0, data_size=1 (2B), no pad/iterate
      g1[1] = 0x04000000;        // tensor_dim0 = 1024 (bits 79:48 low half)
      g1[2] = 0x04000000;        // tensor_dim0 hi=0 | tensor_dim1=1024 low half
      g1[3] = 0x00800000;        // tensor_dim1 hi=0 | tile_dim0=128
      g1[4] = 16;                // tile_dim1=16, tile_dim2=0
      g1[5] = 1024;              // tensor_dim0_stride[31:0] = 1024 elements
      g1[6] = 0;                 // stride hi | tensor_dim1_stride lo
      g1[7] = 0;
      i32x4 z4 = {0, 0, 0, 0};   // groups 2/3 unused (2D tensor)
      i32x8 z8 = {0, 0, 0, 0, 0, 0, 0, 0};
      // 6-arg toolchain variant (clang-23 / therock headers)
      __builtin_amdgcn_tensor_load_to_lds(g0, g1, z4, z4, z8, 0);
      __builtin_amdgcn_s_wait_tensorcnt(0);
    }
#else
    {   // cooperative fallback copy: 256 threads x 8 elements
      int idx = threadIdx.x * 8;
      int rr = idx >> 7, cc = idx & 127;
      const __bf16* s = comb + (size_t)(m0 + rr) * 1024 + kc + cc;
#pragma unroll
      for (int j = 0; j < 8; ++j) shA[rr][cc + j] = s[j];
    }
#endif
    __syncthreads();

    for (int k = 0; k < 128; k += 32) {
      v16bf a  = load_a(&shA[0][0], 128, k, lane);            // LDS (ds_load)
      if (k + 32 < 128)
        __builtin_prefetch(B0 + (long)(kc + k + 32 + lane) * VD_, 0, 0);
      v16bf b0 = load_b(B0,      VD_, 1, kc + k, lane);
      v16bf b1 = load_b(B0 + 16, VD_, 1, kc + k, lane);
      acc0 = wmma_bf16(acc0, a, b0);
      acc1 = wmma_bf16(acc1, a, b1);
    }
  }

  int mrow = m0 + (lane >> 4) * 8;
#pragma unroll
  for (int r = 0; r < 8; ++r) {
    int row = mrow + r;                        // b*32 + t
    int bb = row >> 5, tt = row & 31;
    long base = ((long)tt * BB + bb) * VD_;
    int n0 = nbase + (lane & 15);
    out[base + n0]      = tanhf(acc0[r] + out_b[n0]);
    out[base + n0 + 16] = tanhf(acc1[r] + out_b[n0 + 16]);
  }
}

// ---------------------------------------------------------------------------

extern "C" void kernel_launch(void* const* d_in, const int* in_sizes, int n_in,
                              void* d_out, int out_size, void* d_ws, size_t ws_size,
                              hipStream_t stream) {
  (void)in_sizes; (void)n_in; (void)out_size; (void)ws_size;

  const int*   x       = (const int*)d_in[0];
  const int*   y       = (const int*)d_in[1];
  const float* enc_emb = (const float*)d_in[2];
  const float* dec_emb = (const float*)d_in[3];
  // setup_inputs order per gate n in {i,f,o,c}: enc_Wn, enc_bn, dec_Wn, dec_bn
  const float* encW[4] = { (const float*)d_in[8],  (const float*)d_in[4],
                           (const float*)d_in[12], (const float*)d_in[16] };
  const float* encB[4] = { (const float*)d_in[9],  (const float*)d_in[5],
                           (const float*)d_in[13], (const float*)d_in[17] };
  const float* decW[4] = { (const float*)d_in[10], (const float*)d_in[6],
                           (const float*)d_in[14], (const float*)d_in[18] };
  const float* decB[4] = { (const float*)d_in[11], (const float*)d_in[7],
                           (const float*)d_in[15], (const float*)d_in[19] };
  const float* att_W = (const float*)d_in[20];
  const float* att_b = (const float*)d_in[21];
  const float* out_W = (const float*)d_in[22];
  const float* out_b = (const float*)d_in[23];
  float* out = (float*)d_out;

  // workspace carve-up (~84 MB)
  char* p = (char*)d_ws;
  auto carve = [&](size_t n) { char* r = p; p += (n + 255) & ~(size_t)255; return (void*)r; };
  __bf16* encWT[4]; for (int g = 0; g < 4; ++g) encWT[g] = (__bf16*)carve((size_t)1024 * HH * 2);
  __bf16* decWT[4]; for (int g = 0; g < 4; ++g) decWT[g] = (__bf16*)carve((size_t)1024 * HH * 2);
  __bf16* attWT  = (__bf16*)carve((size_t)HH * HH * 2);
  __bf16* outWT  = (__bf16*)carve((size_t)1024 * VD_ * 2);
  __bf16* ex     = (__bf16*)carve((size_t)SS * BB * EE * 2);
  __bf16* enc_hs = (__bf16*)carve((size_t)SS * BB * HH * 2);
  __bf16* hbuf[2]; float* cbuf[2];
  for (int i = 0; i < 2; ++i) {
    hbuf[i] = (__bf16*)carve((size_t)BB * HH * 2);
    cbuf[i] = (float*)carve((size_t)BB * HH * 4);
  }
  __bf16* Adec   = (__bf16*)carve((size_t)1024 * 1024 * 2);
  __bf16* dec_h  = (__bf16*)carve((size_t)1024 * HH * 2);
  __bf16* comb   = (__bf16*)carve((size_t)1024 * 1024 * 2);
  float*  scores = (float*)carve((size_t)BB * TT * SS * 4);
  __bf16* watt   = (__bf16*)carve((size_t)BB * TT * SS * 2);

  dim3 tb(32, 8);
  for (int g = 0; g < 4; ++g)
    k_conv_transpose<<<dim3(32, 16), tb, 0, stream>>>(encW[g], encWT[g], HH, 1024);
  for (int g = 0; g < 4; ++g)
    k_conv_transpose<<<dim3(32, 16), tb, 0, stream>>>(decW[g], decWT[g], HH, 1024);
  k_conv_transpose<<<dim3(16, 16),   tb, 0, stream>>>(att_W, attWT, HH, HH);
  k_conv_transpose<<<dim3(32, 1000), tb, 0, stream>>>(out_W, outWT, VD_, 1024);

  k_init_state<<<64, 256, 0, stream>>>(hbuf[0], cbuf[0]);
  k_gather_enc<<<SS * BB, 256, 0, stream>>>(x, enc_emb, ex);
  k_gather_dec<<<BB * TT, 256, 0, stream>>>(x, y, dec_emb, Adec);

  // sequential encoder recurrence (stream order gives the dependency)
  for (int t = 0; t < SS; ++t) {
    k_enc_step<<<dim3(32, 2), 128, 0, stream>>>(
        ex + (size_t)t * BB * EE, hbuf[t & 1], cbuf[t & 1],
        encWT[0], encWT[1], encWT[2], encWT[3],
        encB[0], encB[1], encB[2], encB[3],
        hbuf[(t + 1) & 1], cbuf[(t + 1) & 1],
        enc_hs + (size_t)t * BB * HH);
  }
  // after 128 steps final (h,c) land back in buffer 0
  k_fill_dec_h<<<BB * TT, 256, 0, stream>>>(hbuf[0], Adec);

  k_dec_step<<<dim3(32, 64), 128, 0, stream>>>(
      Adec, cbuf[0],
      decWT[0], decWT[1], decWT[2], decWT[3],
      decB[0], decB[1], decB[2], decB[3], dec_h);

  k_att_proj<<<dim3(8, 64), 128, 0, stream>>>(dec_h, attWT, att_b, comb);
  k_scores<<<dim3(2, 2, BB), 128, 0, stream>>>(comb, enc_hs, scores);
  k_softmax<<<BB * TT / 4, 128, 0, stream>>>(scores, watt);
  k_ctx<<<dim3(8, 2, BB), 128, 0, stream>>>(watt, enc_hs, comb);

  k_logits<<<dim3(125, 64), 256, 0, stream>>>(comb, outWT, out_b, out);
}
